// Encoder_35235911697098
// MI455X (gfx1250) — matrix operand order repaired and checked by reading, not compiled, not run on hardware
//
#include <hip/hip_runtime.h>
#include <hip/hip_bf16.h>
#include <math.h>

// ---------------------------------------------------------------------------
// Mamba2 vision encoder for MI455X (gfx1250, wave32).
// GEMMs: v_wmma_f32_16x16x32_bf16, 2x2 register-blocked tiles per wave
// (32x32 output / wave, A/B fragment reuse halves VMEM+cvt per WMMA).
// Scan / convs / BN / norms run in f32 VALU.
// Workspace requirement: ~87 MB (21,692,416 floats).
// ---------------------------------------------------------------------------

typedef __attribute__((ext_vector_type(16))) __bf16 v16bf;
typedef __attribute__((ext_vector_type(8)))  float  v8f;

#define EPSBN 1e-5f

__device__ __forceinline__ float siluf(float x) { return x / (1.f + __expf(-x)); }

__device__ __forceinline__ float bn_apply(float v, const float* __restrict__ p, int C, int c) {
  float g = p[c], be = p[C + c], mu = p[2 * C + c], va = p[3 * C + c];
  return (v - mu) * rsqrtf(va + EPSBN) * g + be;
}

// A fragment (16-bit A 16x32 layout): lane holds K runs [ka, ka+8) and [16+ka, 16+ka+8)
__device__ __forceinline__ v16bf load_afrag(const float* __restrict__ p) {
  float4 f0 = *(const float4*)(p);
  float4 f1 = *(const float4*)(p + 4);
  float4 f2 = *(const float4*)(p + 16);
  float4 f3 = *(const float4*)(p + 20);
  v16bf r;
  r[0]=(__bf16)f0.x;  r[1]=(__bf16)f0.y;  r[2]=(__bf16)f0.z;  r[3]=(__bf16)f0.w;
  r[4]=(__bf16)f1.x;  r[5]=(__bf16)f1.y;  r[6]=(__bf16)f1.z;  r[7]=(__bf16)f1.w;
  r[8]=(__bf16)f2.x;  r[9]=(__bf16)f2.y;  r[10]=(__bf16)f2.z; r[11]=(__bf16)f2.w;
  r[12]=(__bf16)f3.x; r[13]=(__bf16)f3.y; r[14]=(__bf16)f3.z; r[15]=(__bf16)f3.w;
  return r;
}

// B fragment (16-bit B 32x16 layout): lane holds K runs [kb, kb+8) and [16+kb, 16+kb+8)
__device__ __forceinline__ v16bf load_bfrag(const float* __restrict__ p) {
  float4 f0 = *(const float4*)(p);
  float4 f1 = *(const float4*)(p + 4);
  float4 f2 = *(const float4*)(p + 16);
  float4 f3 = *(const float4*)(p + 20);
  v16bf r;
  r[0]=(__bf16)f0.x;  r[1]=(__bf16)f0.y;  r[2]=(__bf16)f0.z;  r[3]=(__bf16)f0.w;
  r[4]=(__bf16)f1.x;  r[5]=(__bf16)f1.y;  r[6]=(__bf16)f1.z;  r[7]=(__bf16)f1.w;
  r[8]=(__bf16)f2.x;  r[9]=(__bf16)f2.y;  r[10]=(__bf16)f2.z; r[11]=(__bf16)f2.w;
  r[12]=(__bf16)f3.x; r[13]=(__bf16)f3.y; r[14]=(__bf16)f3.z; r[15]=(__bf16)f3.w;
  return r;
}

// --------------------------- conv stem -------------------------------------
// conv1: image [8,1,64,1024] -> [8,32,32,512], 7x7 s2 p3, + BN
__global__ __launch_bounds__(256)
void conv1_kernel(const float* __restrict__ img, const float* __restrict__ w,
                  const float* __restrict__ bnp, float* __restrict__ out) {
  int idx = blockIdx.x * 256 + threadIdx.x;          // 8*32*32*512
  int ww = idx & 511, h = (idx >> 9) & 31, c = (idx >> 14) & 31, b = idx >> 19;
  float acc = 0.f;
  #pragma unroll
  for (int kh = 0; kh < 7; kh++) {
    int ih = 2 * h - 3 + kh;
    if (ih < 0 || ih >= 64) continue;
    #pragma unroll
    for (int kw = 0; kw < 7; kw++) {
      int iw = 2 * ww - 3 + kw;
      if (iw < 0 || iw >= 1024) continue;
      acc += img[(b * 64 + ih) * 1024 + iw] * w[c * 49 + kh * 7 + kw];
    }
  }
  out[idx] = bn_apply(acc, bnp, 32, c);
}

// conv2 + downsample residual + relu -> tokens [8,1024,256]
__global__ __launch_bounds__(256)
void conv2_kernel(const float* __restrict__ x1, const float* __restrict__ w,
                  const float* __restrict__ bnp, const float* __restrict__ img,
                  const float* __restrict__ dsw, const float* __restrict__ dsbn,
                  float* __restrict__ tokens) {
  int idx = blockIdx.x * 256 + threadIdx.x;          // 8*64*16*256
  int ww = idx & 255, h = (idx >> 8) & 15, c = (idx >> 12) & 63, b = idx >> 18;
  float acc = 0.f;
  for (int ci = 0; ci < 32; ci++) {
    const float* xc = x1 + ((b * 32 + ci) * 32) * 512;
    const float* wc = w + ((c * 32 + ci) * 49);
    #pragma unroll
    for (int kh = 0; kh < 7; kh++) {
      int ih = 2 * h - 3 + kh;
      if (ih < 0 || ih >= 32) continue;
      #pragma unroll
      for (int kw = 0; kw < 7; kw++) {
        int iw = 2 * ww - 3 + kw;
        if (iw < 0 || iw >= 512) continue;
        acc += xc[ih * 512 + iw] * wc[kh * 7 + kw];
      }
    }
  }
  float v = bn_apply(acc, bnp, 64, c);
  float dv = bn_apply(img[(b * 64 + 4 * h) * 1024 + 4 * ww] * dsw[c], dsbn, 64, c);
  v = fmaxf(v + dv, 0.f);
  tokens[(b * 1024 + c * 16 + h) * 256 + ww] = v;    // channel d = c*16+h
}

// --------------------------- transpose -------------------------------------
// tokens [B, DM, L] -> u [B*L, DM]
__global__ __launch_bounds__(256)
void transpose_kernel(const float* __restrict__ tokens, float* __restrict__ u) {
  int idx = blockIdx.x * 256 + threadIdx.x;          // 2048*1024
  int m = idx >> 10, k = idx & 1023;
  int b = m >> 8, l = m & 255;
  u[idx] = tokens[(b * 1024 + k) * 256 + l];
}

// --------------------------- WMMA GEMM -------------------------------------
// D[M,N] = A[M,K] * W[N,K]^T
// 8 waves/block as 2(M) x 4(N); each wave owns a 32x32 block = 2x2 WMMA tiles.
// Block tile = 64(M) x 128(N).
// mode 0: plain store row-major [M,N]
// mode 1: (v+bias)->BN(N)->relu, row-major [M,N]
// mode 2: (v+bias?)->BN(N), transpose store [B,N,L] + residual (in-place OK)
__global__ __launch_bounds__(256)
void gemm_wmma(const float* __restrict__ A, const float* __restrict__ W,
               float* C, const float* __restrict__ bias,
               const float* __restrict__ bnp, const float* resid,
               int M, int N, int K, int mode) {
  const int lane  = threadIdx.x & 31;
  const int wave  = threadIdx.x >> 5;
  const int ntile = blockIdx.x * 128 + (wave & 3) * 32;
  if (ntile + 32 > N) return;                        // wave-uniform: EXEC stays full
  const int mtile = blockIdx.y * 64 + (wave >> 2) * 32;

  const int am   = lane & 15;
  const int kgrp = lane >> 4;                        // 0/1
  const int ln   = lane & 15;
  const int ka   = kgrp * 8;
  const int kb   = kgrp * 8;

  const float* arow0 = A + (size_t)(mtile + am) * K;
  const float* arow1 = A + (size_t)(mtile + 16 + am) * K;
  const float* wrow0 = W + (size_t)(ntile + ln) * K;
  const float* wrow1 = W + (size_t)(ntile + 16 + ln) * K;

  v8f acc00 = {}, acc01 = {}, acc10 = {}, acc11 = {};
  for (int k0 = 0; k0 < K; k0 += 32) {
    __builtin_prefetch(wrow0 + k0 + 256, 0, 3);      // global_prefetch_b8
    __builtin_prefetch(wrow1 + k0 + 256, 0, 3);
    v16bf a0 = load_afrag(arow0 + k0 + ka);
    v16bf a1 = load_afrag(arow1 + k0 + ka);
    v16bf b0 = load_bfrag(wrow0 + k0 + kb);
    v16bf b1 = load_bfrag(wrow1 + k0 + kb);
    acc00 = __builtin_amdgcn_wmma_f32_16x16x32_bf16(false, a0, false, b0, (short)0, acc00, false, false);
    acc01 = __builtin_amdgcn_wmma_f32_16x16x32_bf16(false, a0, false, b1, (short)0, acc01, false, false);
    acc10 = __builtin_amdgcn_wmma_f32_16x16x32_bf16(false, a1, false, b0, (short)0, acc10, false, false);
    acc11 = __builtin_amdgcn_wmma_f32_16x16x32_bf16(false, a1, false, b1, (short)0, acc11, false, false);
  }

  #pragma unroll
  for (int j = 0; j < 2; j++) {
    const int n = ntile + j * 16 + ln;
    float bsc = 1.f, bsh = 0.f, bia = 0.f;
    if (mode >= 1) {
      float g = bnp[n], be = bnp[N + n], mu = bnp[2 * N + n], va = bnp[3 * N + n];
      bsc = g * rsqrtf(va + EPSBN);
      bsh = be - mu * bsc;
      if (bias) bia = bias[n];
    }
    #pragma unroll
    for (int i = 0; i < 2; i++) {
      v8f acc = (i == 0) ? (j == 0 ? acc00 : acc01) : (j == 0 ? acc10 : acc11);
      #pragma unroll
      for (int r = 0; r < 8; r++) {
        int m = mtile + i * 16 + r + kgrp * 8;       // C/D layout: lanes 16-31 -> M+8
        float v = acc[r];
        if (mode == 0) {
          C[(size_t)m * N + n] = v;
        } else if (mode == 1) {
          v = (v + bia) * bsc + bsh;
          C[(size_t)m * N + n] = fmaxf(v, 0.f);
        } else {
          v = (v + bia) * bsc + bsh;
          int bb = m >> 8, l = m & 255;
          size_t idx = ((size_t)(bb * N + n)) * 256 + l;
          C[idx] = v + resid[idx];
        }
      }
    }
  }
}

// --------------------------- mamba pieces ----------------------------------
// dt = softplus(raw + dt_bias); dA = exp(dt * (-exp(A_log)))
__global__ __launch_bounds__(256)
void dt_kernel(const float* __restrict__ zx, const float* __restrict__ dtb,
               const float* __restrict__ A_log, float* __restrict__ dt,
               float* __restrict__ dA) {
  int idx = blockIdx.x * 256 + threadIdx.x;          // 2048*32
  int m = idx >> 5, h = idx & 31;
  float raw = zx[(size_t)m * 4256 + 4224 + h] + dtb[h];
  float d = (raw > 20.f) ? raw : log1pf(__expf(raw));
  dt[idx] = d;
  dA[idx] = __expf(d * (-__expf(A_log[h])));
}

// causal depthwise conv (DC=4) + bias + silu over xBC cols of zxbcdt
__global__ __launch_bounds__(256)
void conv1d_kernel(const float* __restrict__ zx, const float* __restrict__ cw,
                   const float* __restrict__ cb, float* __restrict__ xS,
                   float* __restrict__ Bs, float* __restrict__ Cs) {
  int idx = blockIdx.x * 256 + threadIdx.x;          // 2048*2176
  int m = idx / 2176, c = idx - m * 2176;
  int b = m >> 8, l = m & 255;
  float acc = cb[c];
  #pragma unroll
  for (int k = 0; k < 4; k++) {
    int ll = l + k - 3;
    if (ll >= 0)
      acc += cw[c * 4 + k] * zx[(size_t)(b * 256 + ll) * 4256 + 2048 + c];
  }
  acc = siluf(acc);
  if (c < 2048)      xS[(size_t)m * 2048 + c]        = acc;
  else if (c < 2112) Bs[(size_t)m * 64 + (c - 2048)] = acc;
  else               Cs[(size_t)m * 64 + (c - 2112)] = acc;
}

// selective state scan: one block per (b,h); state h[64,64] in registers
// (16 f32 per thread), per-step B/C/x staged in LDS; y += D*x skip.
__global__ __launch_bounds__(256)
void scan_kernel(const float* __restrict__ xS, const float* __restrict__ Bs,
                 const float* __restrict__ Cs, const float* __restrict__ dt,
                 const float* __restrict__ dA, const float* __restrict__ Dp,
                 float* __restrict__ y) {
  int bh = blockIdx.x;                               // 256 = B*NH
  int b = bh >> 5, h = bh & 31;
  int tid = threadIdx.x;
  int p = tid >> 2, ng = tid & 3;                    // 4 lanes per p share n-groups
  __shared__ float xh[64], Bsh[64], Csh[64], sc[2];
  float hs[16];
  #pragma unroll
  for (int j = 0; j < 16; j++) hs[j] = 0.f;
  const float Dh = Dp[h];

  for (int l = 0; l < 256; l++) {
    int m = b * 256 + l;
    if (tid < 64)        xh[tid]        = xS[(size_t)m * 2048 + h * 64 + tid];
    else if (tid < 128)  Bsh[tid - 64]  = Bs[(size_t)m * 64 + (tid - 64)];
    else if (tid < 192)  Csh[tid - 128] = Cs[(size_t)m * 64 + (tid - 128)];
    else if (tid == 192) { sc[0] = dt[m * 32 + h]; sc[1] = dA[m * 32 + h]; }
    __syncthreads();
    float dtx = sc[0] * xh[p];
    float da  = sc[1];
    float acc = 0.f;
    #pragma unroll
    for (int j = 0; j < 16; j++) {
      int nn = ng * 16 + j;
      hs[j] = hs[j] * da + dtx * Bsh[nn];
      acc  += hs[j] * Csh[nn];
    }
    acc += __shfl_xor(acc, 1, 32);
    acc += __shfl_xor(acc, 2, 32);
    if (ng == 0) y[(size_t)m * 2048 + h * 64 + p] = acc + Dh * xh[p];
    __syncthreads();
  }
}

// y *= silu(z); gated RMSNorm with norm_w
__global__ __launch_bounds__(256)
void gate_norm_kernel(const float* __restrict__ y, const float* __restrict__ zx,
                      const float* __restrict__ nw, float* __restrict__ yn) {
  int m = blockIdx.x, tid = threadIdx.x;
  __shared__ float red[8];
  float v[8];
  float ss = 0.f;
  #pragma unroll
  for (int i = 0; i < 8; i++) {
    int c = tid + i * 256;
    float val = y[(size_t)m * 2048 + c] * siluf(zx[(size_t)m * 4256 + c]);
    v[i] = val;
    ss += val * val;
  }
  ss += __shfl_xor(ss, 1, 32);  ss += __shfl_xor(ss, 2, 32);
  ss += __shfl_xor(ss, 4, 32);  ss += __shfl_xor(ss, 8, 32);
  ss += __shfl_xor(ss, 16, 32);
  if ((tid & 31) == 0) red[tid >> 5] = ss;
  __syncthreads();
  float tot = red[0] + red[1] + red[2] + red[3] + red[4] + red[5] + red[6] + red[7];
  float rinv = rsqrtf(tot * (1.f / 2048.f) + EPSBN);
  #pragma unroll
  for (int i = 0; i < 8; i++) {
    int c = tid + i * 256;
    yn[(size_t)m * 2048 + c] = v[i] * rinv * nw[c];
  }
}

// --------------------------- launch ----------------------------------------
extern "C" void kernel_launch(void* const* d_in, const int* in_sizes, int n_in,
                              void* d_out, int out_size, void* d_ws, size_t ws_size,
                              hipStream_t stream) {
  (void)in_sizes; (void)n_in; (void)out_size; (void)ws_size;
  const float* image    = (const float*)d_in[0];
  const float* conv1_w  = (const float*)d_in[1];
  const float* bn1p     = (const float*)d_in[2];
  const float* conv2_w  = (const float*)d_in[3];
  const float* bn2p     = (const float*)d_in[4];
  const float* ds_w     = (const float*)d_in[5];
  const float* ds_bn    = (const float*)d_in[6];
  const float* in_proj  = (const float*)d_in[7];
  const float* c1dw     = (const float*)d_in[8];
  const float* c1db     = (const float*)d_in[9];
  const float* dtb      = (const float*)d_in[10];
  const float* A_log    = (const float*)d_in[11];
  const float* Dp       = (const float*)d_in[12];
  const float* norm_w   = (const float*)d_in[13];
  const float* out_proj = (const float*)d_in[14];
  const float* blk_bn   = (const float*)d_in[15];
  const float* ff_w1    = (const float*)d_in[16];
  const float* ff_b1    = (const float*)d_in[17];
  const float* ff_bn1   = (const float*)d_in[18];
  const float* ff_w2    = (const float*)d_in[19];
  const float* ff_b2    = (const float*)d_in[20];
  const float* ff_bn2   = (const float*)d_in[21];

  float* tokens = (float*)d_out;                     // [8,1024,256] lives in d_out
  float* ws = (float*)d_ws;
  // aliased workspace layout (floats): total 21,692,416 (~87 MB)
  float* zxbcdt = ws;                                // 8,716,288  [2048,4256]
  float* xS     = ws + 8716288;                      // 4,194,304  (also yn / tkT)
  float* yb     = ws + 12910592;                     // 4,194,304  (also conv1 out)
  float* ub     = ws + 17104896;                     // 4,194,304  (u / ff hidden)
  float* dtv    = ws + 21299200;                     // 65,536
  float* dAv    = ws + 21364736;                     // 65,536
  float* Bsb    = ws + 21430272;                     // 131,072
  float* Csb    = ws + 21561344;                     // 131,072
  float* c1b = yb;
  float* yn  = xS;
  float* tkT = xS;
  float* ffh = ub;

  dim3 b256(256);

  conv1_kernel<<<16384, b256, 0, stream>>>(image, conv1_w, bn1p, c1b);
  conv2_kernel<<<2048, b256, 0, stream>>>(c1b, conv2_w, bn2p, image, ds_w, ds_bn, tokens);

  for (int i = 0; i < 4; i++) {
    // mamba2: in_proj -> dt/conv1d -> scan -> gated RMSNorm -> out_proj(+BN+res)
    transpose_kernel<<<8192, b256, 0, stream>>>(tokens, ub);
    gemm_wmma<<<dim3(34, 32), b256, 0, stream>>>(ub, in_proj + (size_t)i * 4358144,
        zxbcdt, nullptr, nullptr, nullptr, 2048, 4256, 1024, 0);
    dt_kernel<<<256, b256, 0, stream>>>(zxbcdt, dtb + i * 32, A_log + i * 32, dtv, dAv);
    conv1d_kernel<<<17408, b256, 0, stream>>>(zxbcdt, c1dw + i * 8704, c1db + i * 2176,
        xS, Bsb, Csb);
    scan_kernel<<<256, b256, 0, stream>>>(xS, Bsb, Csb, dtv, dAv, Dp + i * 32, yb);
    gate_norm_kernel<<<2048, b256, 0, stream>>>(yb, zxbcdt, norm_w + i * 2048, yn);
    gemm_wmma<<<dim3(8, 32), b256, 0, stream>>>(yn, out_proj + (size_t)i * 2097152,
        tokens, nullptr, blk_bn + i * 4096, tokens, 2048, 1024, 2048, 2);
    // feed-forward: relu(bn(W1 x + b1)) -> bn(W2 h + b2) + residual
    transpose_kernel<<<8192, b256, 0, stream>>>(tokens, tkT);
    gemm_wmma<<<dim3(16, 32), b256, 0, stream>>>(tkT, ff_w1 + (size_t)i * 2097152,
        ffh, ff_b1 + i * 2048, ff_bn1 + i * 8192, nullptr, 2048, 2048, 1024, 1);
    gemm_wmma<<<dim3(8, 32), b256, 0, stream>>>(ffh, ff_w2 + (size_t)i * 2097152,
        tokens, ff_b2 + i * 1024, ff_bn2 + i * 4096, tokens, 2048, 1024, 2048, 2);
  }
}
